// GroupedQueryAttention_40389872451880
// MI455X (gfx1250) — matrix operand order repaired and checked by reading, not compiled, not run on hardware
//
#include <hip/hip_runtime.h>
#include <hip/hip_bf16.h>
#include <math.h>

// ---------------------------------------------------------------------------
// GQA on gfx1250: all GEMMs via v_wmma_f32_16x16x32_f16 (f16 in, f32 acc).
// Pipeline: convert -> QKV projection GEMMs -> flash-attention per head.
// ---------------------------------------------------------------------------

#define SEQ      2048
#define D_IN     2048
#define NHEADS   32
#define NKV      8
#define HDIM     64
#define DKQ      2048   // NHEADS * HDIM
#define DKV      512    // NKV * HDIM

typedef __attribute__((ext_vector_type(16))) _Float16 v16h;
typedef __attribute__((ext_vector_type(8)))  _Float16 v8h;   // 16 bytes
typedef __attribute__((ext_vector_type(8)))  float    v8f;

union FragU { v16h v; v8h h[2]; };

// A-matrix 16x32 f16 fragment (ISA 7.12.2): lane<16 holds row M=lane,
// K = {0..7, 16..23}; lane>=16 holds row M=lane-16, K = {8..15, 24..31}.
// With a row-major source this is two contiguous 16B loads per lane.
__device__ __forceinline__ v16h load_frag_a(const _Float16* src, int ld, int lane) {
    const int row = lane & 15;
    const _Float16* p = src + row * ld + ((lane < 16) ? 0 : 8);
    FragU f;
    f.h[0] = *(const v8h*)(p);
    f.h[1] = *(const v8h*)(p + 16);
    return f.v;
}

// B-matrix 32x16 f16 fragment: lane<16 holds column N=lane, K={0..7,16..23};
// lane>=16 holds column N=lane-16, K={8..15,24..31}. Source is "Bt" = column-major B
// stored row-major as Bt[n][k] -> two contiguous 16B loads per lane.
__device__ __forceinline__ v16h load_frag_b(const _Float16* src, int ld, int lane) {
    const int row = lane & 15;
    const _Float16* p = src + row * ld + ((lane < 16) ? 0 : 8);
    FragU f;
    f.h[0] = *(const v8h*)(p);
    f.h[1] = *(const v8h*)(p + 16);
    return f.v;
}

__device__ __forceinline__ v8f wmma32(v16h a, v16h b, v8f c) {
    // (neg_a, A, neg_b, B, c_mod, C, reuse_a, reuse_b)
    return __builtin_amdgcn_wmma_f32_16x16x32_f16(false, a, false, b, (short)0, c,
                                                  false, false);
}

// ---------------------------------------------------------------------------
// f32 -> f16 elementwise convert (x), vectorized.
// ---------------------------------------------------------------------------
__global__ void __launch_bounds__(256)
convert_f16_kernel(const float* __restrict__ src, _Float16* __restrict__ dst, int n) {
    int i = (blockIdx.x * 256 + threadIdx.x) * 4;
    if (i + 3 < n) {
        float4 v = *(const float4*)(src + i);
        dst[i + 0] = (_Float16)v.x;
        dst[i + 1] = (_Float16)v.y;
        dst[i + 2] = (_Float16)v.z;
        dst[i + 3] = (_Float16)v.w;
    }
}

// ---------------------------------------------------------------------------
// W[K][N] f32  ->  Wt[N][K] f16  (32x32 LDS tile transpose, coalesced both ways)
// grid = (N/32, K/32), block = 256 flat.
// ---------------------------------------------------------------------------
__global__ void __launch_bounds__(256)
transpose_convert_kernel(const float* __restrict__ W, _Float16* __restrict__ Wt,
                         int K, int N) {
    __shared__ float tile[32][33];
    const int n0 = blockIdx.x * 32;
    const int k0 = blockIdx.y * 32;
    const int tx = threadIdx.x & 31;
    const int ty = threadIdx.x >> 5;   // 0..7
    for (int i = 0; i < 4; ++i) {
        int k = ty * 4 + i;
        tile[k][tx] = W[(size_t)(k0 + k) * N + n0 + tx];
    }
    __syncthreads();
    for (int i = 0; i < 4; ++i) {
        int nr = ty * 4 + i;
        Wt[(size_t)(n0 + nr) * K + k0 + tx] = (_Float16)tile[tx][nr];
    }
}

// ---------------------------------------------------------------------------
// C[f16] = A[f16, MxK row-major] x Bt[f16, NxK row-major (i.e. B col-major)]
// Block tile 64x128 (8 waves, 2x4), wave tile 32x32 (2x2 WMMA), K-step 32.
// TRANS_OUT stores C transposed (used to produce V^T directly).
// ---------------------------------------------------------------------------
template <bool TRANS_OUT>
__global__ void __launch_bounds__(256)
gemm_f16_kernel(const _Float16* __restrict__ A, const _Float16* __restrict__ Bt,
                _Float16* __restrict__ C, int K, int lda, int ldb, int ldc) {
    const int lane = threadIdx.x & 31;
    const int wid  = threadIdx.x >> 5;     // 0..7
    const int m0   = blockIdx.x * 64 + (wid >> 2) * 32;
    const int n0   = blockIdx.y * 128 + (wid & 3) * 32;

    v8f acc[2][2] = {};
    for (int k = 0; k < K; k += 32) {
        v16h a0 = load_frag_a(A + (size_t)m0 * lda + k, lda, lane);
        v16h a1 = load_frag_a(A + (size_t)(m0 + 16) * lda + k, lda, lane);
        v16h b0 = load_frag_b(Bt + (size_t)n0 * ldb + k, ldb, lane);
        v16h b1 = load_frag_b(Bt + (size_t)(n0 + 16) * ldb + k, ldb, lane);
        acc[0][0] = wmma32(a0, b0, acc[0][0]);
        acc[0][1] = wmma32(a0, b1, acc[0][1]);
        acc[1][0] = wmma32(a1, b0, acc[1][0]);
        acc[1][1] = wmma32(a1, b1, acc[1][1]);
    }
    // C/D layout: lane<16 -> rows r0..r0+7 = VGPR idx (+0), col = lane;
    // lane>=16 -> rows +8, col = lane-16.
    const int cn = lane & 15;
    const int r0 = (lane < 16) ? 0 : 8;
    for (int i = 0; i < 2; ++i)
        for (int j = 0; j < 2; ++j)
            for (int r = 0; r < 8; ++r) {
                int row = m0 + i * 16 + r0 + r;
                int col = n0 + j * 16 + cn;
                _Float16 val = (_Float16)acc[i][j][r];
                if (TRANS_OUT) C[(size_t)col * ldc + row] = val;
                else           C[(size_t)row * ldc + col] = val;
            }
}

// ---------------------------------------------------------------------------
// Flash attention, one head x 128 query rows per block. 8 waves x 16 rows.
// Q16: [SEQ][DKQ] f16, K16: [SEQ][DKV] f16, Vt16: [DKV][SEQ] f16 (transposed).
// out: [SEQ][DKQ] f32. Scale = 1/sqrt(NHEADS) per the reference.
// ---------------------------------------------------------------------------
__global__ void __launch_bounds__(256)
gqa_attention_kernel(const _Float16* __restrict__ Q, const _Float16* __restrict__ Kk,
                     const _Float16* __restrict__ Vt, float* __restrict__ out) {
    __shared__ __align__(16) _Float16 Pbuf[8][16 * 128];   // 32 KB, per-wave tiles

    const int lane = threadIdx.x & 31;
    const int wid  = threadIdx.x >> 5;
    const int h    = blockIdx.y;
    const int kvh  = h >> 2;                // n_rep = NHEADS/NKV = 4
    const int qblk = blockIdx.x;
    const int q0   = qblk * 128 + wid * 16; // this wave's 16 query rows
    const int qcol = h * HDIM;
    const int kcol = kvh * HDIM;
    const float scale = 0.17677669529663687f; // 1/sqrt(32)

    const int cn    = lane & 15;
    const int rbase = (lane < 16) ? 0 : 8;

    // Q fragments for the whole kernel (d = 0..63 -> two K=32 chunks)
    v16h qa0 = load_frag_a(Q + (size_t)q0 * DKQ + qcol + 0,  DKQ, lane);
    v16h qa1 = load_frag_a(Q + (size_t)q0 * DKQ + qcol + 32, DKQ, lane);

    v8f o[4] = {};
    float mrow[8], lrow[8];
    for (int r = 0; r < 8; ++r) { mrow[r] = -INFINITY; lrow[r] = 0.f; }

    for (int jb = 0; jb <= qblk; ++jb) {
        const int kb = jb * 128;
        const bool diag = (jb == qblk);

        // ---- S = Q K^T for 8 key tiles of 16 (2 WMMAs each), scale + mask ----
        float p[8][8];    // [key tile][row VGPR]
        for (int t = 0; t < 8; ++t) {
            const _Float16* kt = Kk + (size_t)(kb + t * 16) * DKV + kcol;
            v16h b0 = load_frag_b(kt + 0,  DKV, lane);
            v16h b1 = load_frag_b(kt + 32, DKV, lane);
            v8f s = {};
            s = wmma32(qa0, b0, s);
            s = wmma32(qa1, b1, s);
            for (int r = 0; r < 8; ++r) {
                float v = s[r] * scale;
                if (diag && (kb + t * 16 + cn) > (q0 + rbase + r)) v = -INFINITY;
                p[t][r] = v;
            }
        }

        // ---- online softmax: row max over 128 keys (tiles, then 16-lane group) ----
        float mnew[8], alpha[8];
        for (int r = 0; r < 8; ++r) {
            float m = p[0][r];
            for (int t = 1; t < 8; ++t) m = fmaxf(m, p[t][r]);
            m = fmaxf(m, __shfl_xor(m, 1, 32));
            m = fmaxf(m, __shfl_xor(m, 2, 32));
            m = fmaxf(m, __shfl_xor(m, 4, 32));
            m = fmaxf(m, __shfl_xor(m, 8, 32));
            mnew[r] = fmaxf(mrow[r], m);
            alpha[r] = __expf(mrow[r] - mnew[r]);
        }
        for (int r = 0; r < 8; ++r) {
            float sum = 0.f;
            for (int t = 0; t < 8; ++t) {
                float e = __expf(p[t][r] - mnew[r]);
                p[t][r] = e;
                sum += e;
            }
            sum += __shfl_xor(sum, 1, 32);
            sum += __shfl_xor(sum, 2, 32);
            sum += __shfl_xor(sum, 4, 32);
            sum += __shfl_xor(sum, 8, 32);
            lrow[r] = lrow[r] * alpha[r] + sum;
            mrow[r] = mnew[r];
        }
        for (int dt = 0; dt < 4; ++dt)
            for (int r = 0; r < 8; ++r)
                o[dt][r] *= alpha[r];

        // ---- stage P (16x128) into this wave's LDS tile, row-major ----
        _Float16* pb = &Pbuf[wid][0];
        for (int t = 0; t < 8; ++t)
            for (int r = 0; r < 8; ++r)
                pb[(rbase + r) * 128 + t * 16 + cn] = (_Float16)p[t][r];
        __syncthreads();   // fence: make ds stores visible/ordered before frag reads

        // ---- O += P V : 4 key chunks x 4 d-tiles = 16 WMMAs ----
        for (int kc = 0; kc < 4; ++kc) {
            v16h pa = load_frag_a(pb + kc * 32, 128, lane);
            for (int dt = 0; dt < 4; ++dt) {
                const _Float16* vt = Vt + (size_t)(kcol + dt * 16) * SEQ + kb + kc * 32;
                v16h vb = load_frag_b(vt, SEQ, lane);
                o[dt] = wmma32(pa, vb, o[dt]);
            }
        }
        __syncthreads();   // protect Pbuf before next iteration overwrites
    }

    // ---- epilogue: normalize and store f32 ctx [SEQ][DKQ] ----
    for (int dt = 0; dt < 4; ++dt) {
        float inv;
        for (int r = 0; r < 8; ++r) {
            int row = q0 + rbase + r;
            int col = qcol + dt * 16 + cn;
            inv = 1.0f / lrow[r];
            out[(size_t)row * DKQ + col] = o[dt][r] * inv;
        }
    }
}

// ---------------------------------------------------------------------------
// Host launcher. Workspace layout (32 MB total):
//   [ 0,  8M) x16      f16 [2048][2048]
//   [ 8, 16M) Wq_t     f16 [2048][2048]   (Wt[n][k])
//   [16, 18M) Wk_t     f16 [ 512][2048]
//   [18, 20M) Wv_t     f16 [ 512][2048]
//   [20, 28M) Q16      f16 [2048][2048]
//   [28, 30M) K16      f16 [2048][ 512]
//   [30, 32M) Vt16     f16 [ 512][2048]   (V transposed)
// ---------------------------------------------------------------------------
extern "C" void kernel_launch(void* const* d_in, const int* in_sizes, int n_in,
                              void* d_out, int out_size, void* d_ws, size_t ws_size,
                              hipStream_t stream) {
    (void)in_sizes; (void)n_in; (void)out_size; (void)ws_size;
    const float* x  = (const float*)d_in[0];
    const float* Wq = (const float*)d_in[1];
    const float* Wk = (const float*)d_in[2];
    const float* Wv = (const float*)d_in[3];
    float* out = (float*)d_out;

    char* ws = (char*)d_ws;
    const size_t MB = 1u << 20;
    _Float16* x16  = (_Float16*)(ws + 0 * MB);
    _Float16* Wqt  = (_Float16*)(ws + 8 * MB);
    _Float16* Wkt  = (_Float16*)(ws + 16 * MB);
    _Float16* Wvt  = (_Float16*)(ws + 18 * MB);
    _Float16* Q16  = (_Float16*)(ws + 20 * MB);
    _Float16* K16  = (_Float16*)(ws + 28 * MB);
    _Float16* Vt16 = (_Float16*)(ws + 30 * MB);

    // 1) precision conversion / weight transpose
    convert_f16_kernel<<<(SEQ * D_IN) / 1024, 256, 0, stream>>>(x, x16, SEQ * D_IN);
    transpose_convert_kernel<<<dim3(DKQ / 32, D_IN / 32), 256, 0, stream>>>(Wq, Wqt, D_IN, DKQ);
    transpose_convert_kernel<<<dim3(DKV / 32, D_IN / 32), 256, 0, stream>>>(Wk, Wkt, D_IN, DKV);
    transpose_convert_kernel<<<dim3(DKV / 32, D_IN / 32), 256, 0, stream>>>(Wv, Wvt, D_IN, DKV);

    // 2) projections: Q (row-major), K (row-major), V^T (transposed store)
    gemm_f16_kernel<false><<<dim3(SEQ / 64, DKQ / 128), 256, 0, stream>>>(
        x16, Wqt, Q16, D_IN, D_IN, D_IN, DKQ);
    gemm_f16_kernel<false><<<dim3(SEQ / 64, DKV / 128), 256, 0, stream>>>(
        x16, Wkt, K16, D_IN, D_IN, D_IN, DKV);
    gemm_f16_kernel<true><<<dim3(SEQ / 64, DKV / 128), 256, 0, stream>>>(
        x16, Wvt, Vt16, D_IN, D_IN, D_IN, SEQ);

    // 3) causal GQA flash attention
    gqa_attention_kernel<<<dim3(SEQ / 128, NHEADS), 256, 0, stream>>>(Q16, K16, Vt16, out);
}
